// coGNConvLayer_36129264894716
// MI455X (gfx1250) — matrix operand order repaired and checked by reading, not compiled, not run on hardware
//
#include <hip/hip_runtime.h>
#include <hip/hip_bf16.h>
#include <stdint.h>

#define NN 50000
#define NE 800000
#define ND 128
#define ED 128
#define AH 32

typedef __attribute__((ext_vector_type(16))) __bf16 v16bf;
typedef __attribute__((ext_vector_type(8)))  float  v8f;

// bf16 transposed-weight offsets (ushort elements) inside d_ws
#define OW_EW0 0        // 384x128 -> [128][384]
#define OW_EW1 49152    // 128x128
#define OW_EW2 65536
#define OW_EW3 81920
#define OW_EW4 98304
#define OW_NW0 114688
#define OW_AW0 131072   // 128x32 -> [32][128]
#define WT_ELEMS 135168
#define WT_BYTES (WT_ELEMS * 2)

#define PITCH 136   // halfwords per activation row in LDS (128 + 8 pad: conflict-free b128 A-frag loads)
#define ET 64       // edges per block (4 waves x 16 rows)

// round-to-nearest-even conversion (cold path: weight prep only)
__device__ __forceinline__ unsigned short f2bf(float f) {
  unsigned u = __float_as_uint(f);
  u += 0x7fffu + ((u >> 16) & 1u);
  return (unsigned short)(u >> 16);
}

// hot path: round-half-up bf16, pure dword bit-ops (add/add/lshr/and_or = 4 VALU per pair)
__device__ __forceinline__ unsigned pack_bf16(float a, float b) {
  unsigned ua = __float_as_uint(a) + 0x8000u;
  unsigned ub = __float_as_uint(b) + 0x8000u;
  return (ub & 0xffff0000u) | (ua >> 16);
}
__device__ __forceinline__ unsigned short f2bf1(float x) {
  return (unsigned short)((__float_as_uint(x) + 0x8000u) >> 16);
}

__device__ __forceinline__ float silu_f(float x) { return x / (1.0f + __expf(-x)); }

struct Frag {
  union { v16bf v; unsigned short u[16]; unsigned w[8]; uint4 q[2]; };
};

__device__ __forceinline__ v8f wmma_bf16(const Frag& A, const Frag& B, v8f c) {
  // D = A(16x32 bf16) x B(32x16 bf16) + C(16x16 f32)
  return __builtin_amdgcn_wmma_f32_16x16x32_bf16(false, A.v, false, B.v, (short)0, c, false, false);
}

// load 8 contiguous f32 from global, convert to 4 dwords of packed bf16
__device__ __forceinline__ void ld8f_bf16(const float* __restrict__ p, unsigned* d) {
  float4 a = *(const float4*)p;
  float4 b = *(const float4*)(p + 4);
  d[0] = pack_bf16(a.x, a.y);
  d[1] = pack_bf16(a.z, a.w);
  d[2] = pack_bf16(b.x, b.y);
  d[3] = pack_bf16(b.z, b.w);
}

__device__ __forceinline__ void atomicMaxF(float* a, float v) {
  if (v >= 0.0f) atomicMax((int*)a, __float_as_int(v));
  else           atomicMin((unsigned int*)a, __float_as_uint(v));
}

// 16xK (A in LDS, row-major, PITCH halfwords) @ Kx(NT*16) (Wt bf16 [n][k]) + bias
template <int NT>
__device__ __forceinline__ void gemmK128_lds(const unsigned short* __restrict__ bufIn,
                                             const unsigned short* __restrict__ w,
                                             const float* __restrict__ bias,
                                             int l15, int half, v8f* acc) {
#pragma unroll
  for (int nt = 0; nt < NT; ++nt) {
    float bv = bias[nt * 16 + l15];
#pragma unroll
    for (int r = 0; r < 8; ++r) acc[nt][r] = bv;
  }
#pragma unroll
  for (int ks = 0; ks < 4; ++ks) {
    const int k0 = ks * 32;
    Frag A;  // lane = row l15; half 0: K=k0+0..7 & k0+16..23, half 1: +8
    A.q[0] = *(const uint4*)(bufIn + l15 * PITCH + k0 + half * 8);
    A.q[1] = *(const uint4*)(bufIn + l15 * PITCH + k0 + 16 + half * 8);
#pragma unroll
    for (int nt = 0; nt < NT; ++nt) {
      Frag B;  // lane = col l15; half h holds K = k0+h*8..+7 & k0+16+h*8..+7 (matches A)
      const unsigned short* wp = w + (nt * 16 + l15) * 128 + k0 + half * 8;
      B.q[0] = *(const uint4*)wp;
      B.q[1] = *(const uint4*)(wp + 16);
      acc[nt] = wmma_bf16(A, B, acc[nt]);
    }
  }
}

// silu epilogue: bf16 -> LDS out buffer; optional f32 -> global (gout = row-base pointer)
__device__ __forceinline__ void epi8(v8f* acc, unsigned short* outb, int l15, int half,
                                     float* __restrict__ gout) {
#pragma unroll
  for (int nt = 0; nt < 8; ++nt) {
#pragma unroll
    for (int r = 0; r < 8; ++r) {
      float x = silu_f(acc[nt][r]);
      outb[(half * 8 + r) * PITCH + nt * 16 + l15] = f2bf1(x);
      if (gout) gout[(size_t)(half * 8 + r) * ED + nt * 16 + l15] = x;
    }
  }
}

// ---------------- kernel 1: weights -> bf16, transposed [n][k] ----------------
__global__ void prep_weights(const float* __restrict__ ew0, const float* __restrict__ ew1,
                             const float* __restrict__ ew2, const float* __restrict__ ew3,
                             const float* __restrict__ ew4, const float* __restrict__ nw0,
                             const float* __restrict__ aw0, unsigned short* __restrict__ wt) {
  int i = blockIdx.x * 256 + threadIdx.x;
  if (i >= WT_ELEMS) return;
  const float* src; int K, N, rel;
  if      (i < OW_EW1) { src = ew0; K = 384; N = 128; rel = i - OW_EW0; }
  else if (i < OW_EW2) { src = ew1; K = 128; N = 128; rel = i - OW_EW1; }
  else if (i < OW_EW3) { src = ew2; K = 128; N = 128; rel = i - OW_EW2; }
  else if (i < OW_EW4) { src = ew3; K = 128; N = 128; rel = i - OW_EW3; }
  else if (i < OW_NW0) { src = ew4; K = 128; N = 128; rel = i - OW_EW4; }
  else if (i < OW_AW0) { src = nw0; K = 128; N = 128; rel = i - OW_NW0; }
  else                 { src = aw0; K = 128; N = 32;  rel = i - OW_AW0; }
  int n = rel / K, k = rel % K;
  wt[i] = f2bf(src[(size_t)k * N + n]);   // Wt[n][k] = W[k][n]
}

// ---------------- kernel 2: init softmax scratch ----------------
__global__ void init_ws(float* __restrict__ seg_max, float* __restrict__ seg_sum,
                        float* __restrict__ agg) {
  int i = blockIdx.x * 256 + threadIdx.x;
  if (i < NN) {
    ((unsigned*)seg_max)[i] = 0xFF800000u;  // -inf
    seg_sum[i] = 0.0f;
  }
  if (i < NN * ND) agg[i] = 0.0f;
}

// ---------------- kernel 3: edge MLP (5 layers) + attention scores ----------------
__global__ __launch_bounds__(128) void edge_mlp_attn(
    const float* __restrict__ node_feats, const float* __restrict__ edge_feats,
    const int* __restrict__ src, const int* __restrict__ dst,
    const unsigned short* __restrict__ wt,
    const float* __restrict__ eb0, const float* __restrict__ eb1, const float* __restrict__ eb2,
    const float* __restrict__ eb3, const float* __restrict__ eb4,
    const float* __restrict__ ab0, const float* __restrict__ aw1, const float* __restrict__ ab1,
    float* __restrict__ eout, float* __restrict__ scores, float* __restrict__ seg_max) {
  __shared__ unsigned short smem[4 * 2 * 16 * PITCH];  // 34,816 B; per-wave ping/pong
  const int tid = threadIdx.x;
  const int wave = tid >> 5, lane = tid & 31;
  const int l15 = lane & 15, half = lane >> 4;
  const int eBase = blockIdx.x * ET + wave * 16;
  const int eA = eBase + l15;  // this lane's A-row edge

  // warm L2/WGP$ with the (block-shared) bf16 weight working set
  for (int off = tid * 256; off < WT_BYTES; off += 128 * 256)
    __builtin_prefetch((const char*)wt + off, 0, 1);

  // concat [e | v_src | v_dst] segment base pointers (f32, row for this lane)
  const float* P[3];
  P[0] = edge_feats + (size_t)eA * ED;
  P[1] = node_feats + (size_t)src[eA] * ND;
  P[2] = node_feats + (size_t)dst[eA] * ND;

  unsigned short* bufA = smem + wave * (2 * 16 * PITCH);
  unsigned short* bufB = bufA + 16 * PITCH;

  v8f acc[8];

  // ---- layer 0: K = 384, A gathered from global and converted in-register ----
#pragma unroll
  for (int nt = 0; nt < 8; ++nt) {
    float bv = eb0[nt * 16 + l15];
#pragma unroll
    for (int r = 0; r < 8; ++r) acc[nt][r] = bv;
  }
#pragma unroll
  for (int ks = 0; ks < 12; ++ks) {
    const int k0 = ks * 32;
    const float* pa = P[k0 >> 7] + (k0 & 127) + half * 8;
    Frag A;
    ld8f_bf16(pa, A.w);
    ld8f_bf16(pa + 16, A.w + 4);
#pragma unroll
    for (int nt = 0; nt < 8; ++nt) {
      Frag B;
      const unsigned short* wp = wt + OW_EW0 + (nt * 16 + l15) * 384 + k0 + half * 8;
      B.q[0] = *(const uint4*)wp;
      B.q[1] = *(const uint4*)(wp + 16);
      acc[nt] = wmma_bf16(A, B, acc[nt]);
    }
  }
  epi8(acc, bufB, l15, half, nullptr);  // L0 -> bufB

  // ---- layers 1..4: wave-private LDS ping-pong, no barriers needed ----
  gemmK128_lds<8>(bufB, wt + OW_EW1, eb1, l15, half, acc);
  epi8(acc, bufA, l15, half, nullptr);
  gemmK128_lds<8>(bufA, wt + OW_EW2, eb2, l15, half, acc);
  epi8(acc, bufB, l15, half, nullptr);
  gemmK128_lds<8>(bufB, wt + OW_EW3, eb3, l15, half, acc);
  epi8(acc, bufA, l15, half, nullptr);
  gemmK128_lds<8>(bufA, wt + OW_EW4, eb4, l15, half, acc);
  epi8(acc, bufB, l15, half, eout + (size_t)eBase * ED);  // e' -> d_out (f32) + bufB (bf16)

  // ---- attention: h = silu(e' @ aw0 + ab0) [16x32], then s = h @ aw1 + ab1 ----
  v8f hacc[2];
  gemmK128_lds<2>(bufB, wt + OW_AW0, ab0, l15, half, hacc);
  float* hb = (float*)bufA;  // bufA is free now; 16x32 f32 = 2 KB per wave
#pragma unroll
  for (int nt = 0; nt < 2; ++nt) {
#pragma unroll
    for (int r = 0; r < 8; ++r)
      hb[(half * 8 + r) * AH + nt * 16 + l15] = silu_f(hacc[nt][r]);
  }
  if (lane < 16) {
    const float* hr = hb + lane * AH;
    float s = ab1[0];
#pragma unroll
    for (int j = 0; j < AH; ++j) s += hr[j] * aw1[j];
    int e2 = eBase + lane;
    scores[e2] = s;
    atomicMaxF(&seg_max[dst[e2]], s);
  }
}

// ---------------- kernel 4: exp(s - max), segment sum ----------------
__global__ void softmax_pass2(float* __restrict__ sc, const float* __restrict__ seg_max,
                              float* __restrict__ seg_sum, const int* __restrict__ dst) {
  int e = blockIdx.x * 256 + threadIdx.x;
  if (e >= NE) return;
  int d = dst[e];
  float ex = __expf(sc[e] - seg_max[d]);
  sc[e] = ex;
  atomicAdd(&seg_sum[d], ex);
}

// ---------------- kernel 5: weighted aggregation into nodes ----------------
__global__ void aggregate_pass3(const float* __restrict__ sc, const float* __restrict__ seg_sum,
                                const int* __restrict__ dst, const float* __restrict__ eprime,
                                float* __restrict__ agg) {
  int t = blockIdx.x * 256 + threadIdx.x;  // 32 lanes per edge, 4 feats per lane
  int e = t >> 5;
  if (e >= NE) return;
  int j = (t & 31) * 4;
  int d = dst[e];
  float w = sc[e] / (seg_sum[d] + 1e-8f);
  float4 v = *(const float4*)(eprime + (size_t)e * ED + j);
  float* ap = agg + (size_t)d * ED + j;
  atomicAdd(ap + 0, v.x * w);
  atomicAdd(ap + 1, v.y * w);
  atomicAdd(ap + 2, v.z * w);
  atomicAdd(ap + 3, v.w * w);
}

// ---------------- kernel 6: node update (WMMA) + residual ----------------
__global__ __launch_bounds__(128) void node_update(const float* __restrict__ agg,
                                                   const unsigned short* __restrict__ wt,
                                                   const float* __restrict__ nb0,
                                                   const float* __restrict__ node_feats,
                                                   float* __restrict__ nout) {
  const int tid = threadIdx.x;
  const int wave = tid >> 5, lane = tid & 31;
  const int l15 = lane & 15, half = lane >> 4;
  const int rBase = blockIdx.x * 64 + wave * 16;
  int rA = rBase + l15;
  if (rA >= NN) rA = NN - 1;  // clamp; out-of-range rows computed but not stored
  const float* pa = agg + (size_t)rA * ED;

  // warm cache with nw0 (32 KB bf16)
  for (int off = tid * 256; off < 128 * 128 * 2; off += 128 * 256)
    __builtin_prefetch((const char*)(wt + OW_NW0) + off, 0, 1);

  v8f acc[8];
#pragma unroll
  for (int nt = 0; nt < 8; ++nt) {
    float bv = nb0[nt * 16 + l15];
#pragma unroll
    for (int r = 0; r < 8; ++r) acc[nt][r] = bv;
  }
#pragma unroll
  for (int ks = 0; ks < 4; ++ks) {
    const int k0 = ks * 32;
    Frag A;
    ld8f_bf16(pa + k0 + half * 8, A.w);
    ld8f_bf16(pa + k0 + 16 + half * 8, A.w + 4);
#pragma unroll
    for (int nt = 0; nt < 8; ++nt) {
      Frag B;
      const unsigned short* wp = wt + OW_NW0 + (nt * 16 + l15) * 128 + k0 + half * 8;
      B.q[0] = *(const uint4*)wp;
      B.q[1] = *(const uint4*)(wp + 16);
      acc[nt] = wmma_bf16(A, B, acc[nt]);
    }
  }
#pragma unroll
  for (int nt = 0; nt < 8; ++nt) {
#pragma unroll
    for (int r = 0; r < 8; ++r) {
      int m = rBase + half * 8 + r;
      if (m < NN) {
        int n = nt * 16 + l15;
        nout[(size_t)m * ND + n] = silu_f(acc[nt][r]) + node_feats[(size_t)m * ND + n];
      }
    }
  }
}

extern "C" void kernel_launch(void* const* d_in, const int* in_sizes, int n_in,
                              void* d_out, int out_size, void* d_ws, size_t ws_size,
                              hipStream_t stream) {
  const float* node_feats = (const float*)d_in[0];
  const float* edge_feats = (const float*)d_in[1];
  const int*   src        = (const int*)d_in[2];
  const int*   dst        = (const int*)d_in[3];
  const float* ew0 = (const float*)d_in[4];  const float* eb0 = (const float*)d_in[5];
  const float* ew1 = (const float*)d_in[6];  const float* eb1 = (const float*)d_in[7];
  const float* ew2 = (const float*)d_in[8];  const float* eb2 = (const float*)d_in[9];
  const float* ew3 = (const float*)d_in[10]; const float* eb3 = (const float*)d_in[11];
  const float* ew4 = (const float*)d_in[12]; const float* eb4 = (const float*)d_in[13];
  const float* nw0 = (const float*)d_in[14]; const float* nb0 = (const float*)d_in[15];
  const float* aw0 = (const float*)d_in[16]; const float* ab0 = (const float*)d_in[17];
  const float* aw1 = (const float*)d_in[18]; const float* ab1 = (const float*)d_in[19];

  float* nout = (float*)d_out;                 // [NN, ND]
  float* eout = nout + (size_t)NN * ND;        // [NE, ED]

  unsigned short* wt = (unsigned short*)d_ws;  // bf16 transposed weights
  float* fws     = (float*)((char*)d_ws + WT_BYTES);
  float* scores  = fws;                        // [NE] (reused as exp_s)
  float* seg_max = scores + NE;                // [NN]
  float* seg_sum = seg_max + NN;               // [NN]
  float* agg     = seg_sum + NN;               // [NN, ND]

  prep_weights<<<(WT_ELEMS + 255) / 256, 256, 0, stream>>>(ew0, ew1, ew2, ew3, ew4, nw0, aw0, wt);
  init_ws<<<(NN * ND + 255) / 256, 256, 0, stream>>>(seg_max, seg_sum, agg);
  edge_mlp_attn<<<NE / ET, 128, 0, stream>>>(node_feats, edge_feats, src, dst, wt,
                                             eb0, eb1, eb2, eb3, eb4, ab0, aw1, ab1,
                                             eout, scores, seg_max);
  softmax_pass2<<<(NE + 255) / 256, 256, 0, stream>>>(scores, seg_max, seg_sum, dst);
  aggregate_pass3<<<(NE * 32) / 256, 256, 0, stream>>>(scores, seg_sum, dst, eout, agg);
  node_update<<<(NN + 63) / 64, 128, 0, stream>>>(agg, wt, nb0, node_feats, nout);
}
